// MoEAttention_10952166605243
// MI455X (gfx1250) — matrix operand order repaired and checked by reading, not compiled, not run on hardware
//
#include <hip/hip_runtime.h>

#define T_TOK 4096
#define C_DIM 768
#define E_EXP 24
#define D_DIM 64
#define H_TOP 12
#define N_SEQ 2048

typedef __bf16 bf16_t;
typedef __attribute__((ext_vector_type(16))) __bf16 v16bf;
typedef __attribute__((ext_vector_type(8)))  __bf16 v8bf;
typedef __attribute__((ext_vector_type(8)))  float  v8f;

static __device__ __forceinline__ v8f vzero8() {
  v8f v;
#pragma unroll
  for (int i = 0; i < 8; ++i) v[i] = 0.0f;
  return v;
}

static __device__ __forceinline__ v8f wmma_bf16(v16bf a, v16bf b, v8f c) {
  // v_wmma_f32_16x16x32_bf16  D = A(16x32) * B(32x16) + C(16x16 f32)
  return __builtin_amdgcn_wmma_f32_16x16x32_bf16(false, a, false, b, (short)0, c,
                                                 false, false);
}

// A fragment (16x32 bf16). rowp points at A[M][k_window_base] for THIS lane's
// row M = lane&15.  elems 0..7 = K 8*half+{0..7}, elems 8..15 = K 16+8*half+{0..7}.
static __device__ __forceinline__ v16bf load_a_frag(const bf16_t* rowp, int half) {
  v8bf lo = *(const v8bf*)(rowp + 8 * half);
  v8bf hi = *(const v8bf*)(rowp + 16 + 8 * half);
  v16bf r;
#pragma unroll
  for (int i = 0; i < 8; ++i) { r[i] = lo[i]; r[i + 8] = hi[i]; }
  return r;
}

// B fragment (32x16 bf16) from column-major storage (B^T rows). colp points at
// BT[N][k_window_base] for THIS lane's column N = lane&15. elems 0..7 = K 8*half+{0..7}, elems 8..15 = K 16+8*half+{0..7}.
static __device__ __forceinline__ v16bf load_b_frag(const bf16_t* colp, int half) {
  v8bf lo = *(const v8bf*)(colp + 8 * half);
  v8bf hi = *(const v8bf*)(colp + 16 + 8 * half);
  v16bf r;
#pragma unroll
  for (int i = 0; i < 8; ++i) { r[i] = lo[i]; r[i + 8] = hi[i]; }
  return r;
}

static __device__ __forceinline__ float half_max(float v) {
  v = fmaxf(v, __shfl_xor(v, 1, 32));
  v = fmaxf(v, __shfl_xor(v, 2, 32));
  v = fmaxf(v, __shfl_xor(v, 4, 32));
  v = fmaxf(v, __shfl_xor(v, 8, 32));
  return v;
}
static __device__ __forceinline__ float half_sum(float v) {
  v += __shfl_xor(v, 1, 32);
  v += __shfl_xor(v, 2, 32);
  v += __shfl_xor(v, 4, 32);
  v += __shfl_xor(v, 8, 32);
  return v;
}

// ---------------------------------------------------------------- init / convert
__global__ void zero_kernel(float* out, int* counts, float* p_sum, float* z_acc) {
  size_t i = (size_t)blockIdx.x * 256 + threadIdx.x;
  if (i < (size_t)T_TOK * C_DIM) out[i] = 0.0f;
  if (i < E_EXP) { counts[i] = 0; p_sum[i] = 0.0f; }
  if (i == 0) *z_acc = 0.0f;
}

__global__ void cvt_x_kernel(const float* __restrict__ x, bf16_t* __restrict__ xb) {
  size_t i = (size_t)blockIdx.x * 256 + threadIdx.x;
  xb[i] = (bf16_t)x[i];
}

__global__ void cvt_wkv_kernel(const float* __restrict__ w, bf16_t* __restrict__ o) {
  // W_kv [C][2D] -> wkvT [2D][C]
  int i = blockIdx.x * 256 + threadIdx.x;
  int j = i / C_DIM, c = i % C_DIM;
  o[i] = (bf16_t)w[c * (2 * D_DIM) + j];
}

__global__ void cvt_win_kernel(const float* __restrict__ w, bf16_t* __restrict__ o) {
  // W_in [E][C][D] -> winT [E][D][C]
  int i = blockIdx.x * 256 + threadIdx.x;
  int e = i / (D_DIM * C_DIM), r = i % (D_DIM * C_DIM);
  int d = r / C_DIM, c = r % C_DIM;
  o[i] = (bf16_t)w[((size_t)e * C_DIM + c) * D_DIM + d];
}

__global__ void cvt_wout_kernel(const float* __restrict__ w, bf16_t* __restrict__ o) {
  // W_out [E][D][C] -> woutT [E][C][D]
  int i = blockIdx.x * 256 + threadIdx.x;
  int e = i / (C_DIM * D_DIM), r = i % (C_DIM * D_DIM);
  int c = r / D_DIM, d = r % D_DIM;
  o[i] = (bf16_t)w[((size_t)e * D_DIM + d) * C_DIM + c];
}

// ---------------------------------------------------------------- gating (wave/token)
__global__ __launch_bounds__(128) void gate_kernel(
    const float* __restrict__ x, const float* __restrict__ wg, float* __restrict__ gates,
    int* __restrict__ entries, int* __restrict__ counts, float* __restrict__ p_sum,
    float* __restrict__ z_acc) {
  const int lane = threadIdx.x & 31;
  const int wave = threadIdx.x >> 5;
  const int t = blockIdx.x * 4 + wave;

  float xv[24];
#pragma unroll
  for (int i = 0; i < 24; ++i) xv[i] = x[(size_t)t * C_DIM + lane + 32 * i];

  float lg[E_EXP];
#pragma unroll
  for (int e = 0; e < E_EXP; ++e) {
    float s = 0.0f;
#pragma unroll
    for (int i = 0; i < 24; ++i) s += xv[i] * wg[(lane + 32 * i) * E_EXP + e];
    s += __shfl_xor(s, 1, 32);
    s += __shfl_xor(s, 2, 32);
    s += __shfl_xor(s, 4, 32);
    s += __shfl_xor(s, 8, 32);
    s += __shfl_xor(s, 16, 32);
    lg[e] = s;  // all lanes hold the full logit
  }

  float m = lg[0];
#pragma unroll
  for (int e = 1; e < E_EXP; ++e) m = fmaxf(m, lg[e]);
  float pr[E_EXP];
  float sum = 0.0f;
#pragma unroll
  for (int e = 0; e < E_EXP; ++e) { pr[e] = __expf(lg[e] - m); sum += pr[e]; }
  float inv = 1.0f / sum;
#pragma unroll
  for (int e = 0; e < E_EXP; ++e) pr[e] *= inv;
  float lse = m + __logf(sum);

  // top-12 (ties -> lowest index, matching lax.top_k)
  int sel[H_TOP];
  float tp[H_TOP];
  bool used[E_EXP];
#pragma unroll
  for (int e = 0; e < E_EXP; ++e) used[e] = false;
#pragma unroll
  for (int h = 0; h < H_TOP; ++h) {
    int best = 0;
    float bv = -1.0f;
#pragma unroll
    for (int e = 0; e < E_EXP; ++e)
      if (!used[e] && pr[e] > bv) { bv = pr[e]; best = e; }
    used[best] = true;
    sel[h] = best;
    tp[h] = bv;
  }
  float gs = 0.0f;
#pragma unroll
  for (int h = 0; h < H_TOP; ++h) gs += tp[h];
  float ginv = 1.0f / (gs + 1e-6f);

#pragma unroll
  for (int h = 0; h < H_TOP; ++h) {
    if (lane == h) {
      int e = sel[h];
      int pos = atomicAdd(&counts[e], 1);
      entries[e * T_TOK + pos] = (t << 4) | h;
      gates[t * H_TOP + h] = tp[h] * ginv;
    }
  }
  if (lane == 0) {
#pragma unroll
    for (int e = 0; e < E_EXP; ++e) unsafeAtomicAdd(&p_sum[e], pr[e]);
    unsafeAtomicAdd(z_acc, lse * lse);
  }
}

__global__ void aux_kernel(const float* p_sum, const int* counts, const float* z_acc,
                           float* out_aux) {
  if (threadIdx.x == 0 && blockIdx.x == 0) {
    float tp = 0.0f, tf = 0.0f;
    for (int e = 0; e < E_EXP; ++e) { tp += p_sum[e]; tf += (float)counts[e]; }
    float sw = 0.0f;
    for (int e = 0; e < E_EXP; ++e)
      sw += (p_sum[e] / tp) * ((float)counts[e] / tf);
    sw *= (float)E_EXP;
    float z = *z_acc / (float)T_TOK;
    *out_aux = 0.1f * sw + 0.001f * z;
  }
}

// ---------------------------------------------------------------- KV projection (WMMA)
__global__ __launch_bounds__(32) void kv_kernel(
    const bf16_t* __restrict__ xb, const bf16_t* __restrict__ wkvT,
    const float* __restrict__ b_kv, bf16_t* __restrict__ kout, bf16_t* __restrict__ vtout) {
  const int lane = threadIdx.x;
  const int half = lane >> 4, n16 = lane & 15;
  const int mt = blockIdx.x >> 3;   // 256 token tiles
  const int nt = blockIdx.x & 7;    // 8 col tiles of 2D=128
  const bf16_t* arow = xb + (size_t)(mt * 16 + n16) * C_DIM;
  const bf16_t* bcol = wkvT + (size_t)(nt * 16 + n16) * C_DIM;
  v8f c = vzero8();
  for (int kb = 0; kb < C_DIM; kb += 32)
    c = wmma_bf16(load_a_frag(arow + kb, half), load_b_frag(bcol + kb, half), c);
#pragma unroll
  for (int r = 0; r < 8; ++r) {
    int tok = mt * 16 + r + 8 * half;
    int col = nt * 16 + n16;  // 0..127
    float v = c[r] + b_kv[col];
    if (col < D_DIM) {
      kout[(size_t)tok * D_DIM + col] = (bf16_t)v;
    } else {
      int bb = tok >> 11, nn = tok & (N_SEQ - 1), d = col - D_DIM;
      vtout[((size_t)bb * D_DIM + d) * N_SEQ + nn] = (bf16_t)v;
    }
  }
}

// ---------------------------------------------------------------- Q grouped GEMM (WMMA)
__global__ __launch_bounds__(32) void qproj_kernel(
    const bf16_t* __restrict__ xb, const bf16_t* __restrict__ winT,
    const float* __restrict__ b_in, const int* __restrict__ entries,
    const int* __restrict__ counts, bf16_t* __restrict__ qout) {
  const int e = blockIdx.x;
  const int base = blockIdx.y * 16;
  const int cnt = counts[e];
  if (base >= cnt) return;
  const int lane = threadIdx.x, half = lane >> 4, n16 = lane & 15;
  int ridx = base + n16;
  if (ridx >= cnt) ridx = cnt - 1;
  const int ent = entries[e * T_TOK + ridx];
  const int t = ent >> 4, slot = ent & 15;
  const bf16_t* arow = xb + (size_t)t * C_DIM;
  const bf16_t* wbase = winT + (size_t)e * D_DIM * C_DIM;
  v8f acc[4];
#pragma unroll
  for (int f = 0; f < 4; ++f) acc[f] = vzero8();
  for (int kb = 0; kb < C_DIM; kb += 32) {
    v16bf a = load_a_frag(arow + kb, half);
#pragma unroll
    for (int f = 0; f < 4; ++f) {
      const bf16_t* bcol = wbase + (size_t)(f * 16 + n16) * C_DIM + kb;
      acc[f] = wmma_bf16(a, load_b_frag(bcol, half), acc[f]);
    }
  }
#pragma unroll
  for (int r = 0; r < 8; ++r) {
    const int M = r + 8 * half;
    const int tS = __shfl(t, M, 32);
    const int sS = __shfl(slot, M, 32);
    if (base + M < cnt) {
      size_t qb = ((size_t)tS * H_TOP + sS) * D_DIM;
#pragma unroll
      for (int f = 0; f < 4; ++f) {
        int d = f * 16 + n16;
        qout[qb + d] = (bf16_t)((acc[f][r] + b_in[e * D_DIM + d]) * 0.125f);
      }
    }
  }
}

// ---------------------------------------------------------------- flash attention (WMMA)
__global__ __launch_bounds__(128) void attn_kernel(
    const bf16_t* __restrict__ q, const bf16_t* __restrict__ kmat,
    const bf16_t* __restrict__ vt, bf16_t* __restrict__ o) {
  __shared__ bf16_t pbuf[4][16 * 32];
  const int lane = threadIdx.x & 31;
  const int wave = threadIdx.x >> 5;
  const int half = lane >> 4, n16 = lane & 15;
  const int qb = blockIdx.x * 4 + wave;
  const int ib = qb & 127;
  const int h = (qb >> 7) % H_TOP;
  const int b = qb / (128 * H_TOP);

  const bf16_t* qrow =
      q + ((size_t)((b * N_SEQ + ib * 16 + n16) * H_TOP + h)) * D_DIM;
  const v16bf aq0 = load_a_frag(qrow + 0, half);
  const v16bf aq1 = load_a_frag(qrow + 32, half);
  const bf16_t* kbase = kmat + (size_t)b * N_SEQ * D_DIM;
  const bf16_t* vbase = vt + (size_t)b * D_DIM * N_SEQ;

  v8f acc[4];
#pragma unroll
  for (int f = 0; f < 4; ++f) acc[f] = vzero8();
  float m_i[8], l_i[8];
#pragma unroll
  for (int r = 0; r < 8; ++r) { m_i[r] = -1e30f; l_i[r] = 0.0f; }

  for (int j = 0; j < N_SEQ; j += 32) {
    v8f s0 = vzero8(), s1 = vzero8();
    const bf16_t* kcol0 = kbase + (size_t)(j + n16) * D_DIM;
    const bf16_t* kcol1 = kbase + (size_t)(j + 16 + n16) * D_DIM;
    s0 = wmma_bf16(aq0, load_b_frag(kcol0 + 0, half), s0);
    s0 = wmma_bf16(aq1, load_b_frag(kcol0 + 32, half), s0);
    s1 = wmma_bf16(aq0, load_b_frag(kcol1 + 0, half), s1);
    s1 = wmma_bf16(aq1, load_b_frag(kcol1 + 32, half), s1);

    float p0[8], p1[8];
#pragma unroll
    for (int r = 0; r < 8; ++r) {
      float mx = half_max(fmaxf(s0[r], s1[r]));
      float mnew = fmaxf(m_i[r], mx);
      float sc = __expf(m_i[r] - mnew);
      p0[r] = __expf(s0[r] - mnew);
      p1[r] = __expf(s1[r] - mnew);
      float rs = half_sum(p0[r] + p1[r]);
      l_i[r] = l_i[r] * sc + rs;
      m_i[r] = mnew;
#pragma unroll
      for (int f = 0; f < 4; ++f) acc[f][r] *= sc;
    }

    // C-layout P -> LDS -> A-layout fragment
#pragma unroll
    for (int r = 0; r < 8; ++r) {
      int row = r + 8 * half;
      pbuf[wave][row * 32 + n16] = (bf16_t)p0[r];
      pbuf[wave][row * 32 + 16 + n16] = (bf16_t)p1[r];
    }
    __syncthreads();
    v16bf ap = load_a_frag(&pbuf[wave][n16 * 32], half);
    __syncthreads();

#pragma unroll
    for (int f = 0; f < 4; ++f) {
      const bf16_t* vcol = vbase + (size_t)(f * 16 + n16) * N_SEQ + j;
      acc[f] = wmma_bf16(ap, load_b_frag(vcol, half), acc[f]);
    }
  }

#pragma unroll
  for (int r = 0; r < 8; ++r) {
    int row = r + 8 * half;
    float invl = 1.0f / l_i[r];
    size_t ob = ((size_t)((b * N_SEQ + ib * 16 + row) * H_TOP + h)) * D_DIM;
#pragma unroll
    for (int f = 0; f < 4; ++f) o[ob + f * 16 + n16] = (bf16_t)(acc[f][r] * invl);
  }
}

// ---------------------------------------------------------------- output grouped GEMM
__global__ __launch_bounds__(128) void oproj_kernel(
    const bf16_t* __restrict__ ob, const bf16_t* __restrict__ woutT,
    const float* __restrict__ b_out, const float* __restrict__ gates,
    const int* __restrict__ entries, const int* __restrict__ counts,
    float* __restrict__ out) {
  const int e = blockIdx.x;
  const int base = blockIdx.y * 16;
  const int cnt = counts[e];
  if (base >= cnt) return;
  const int lane = threadIdx.x & 31, wave = threadIdx.x >> 5;
  const int half = lane >> 4, n16 = lane & 15;
  int ridx = base + n16;
  if (ridx >= cnt) ridx = cnt - 1;
  const int ent = entries[e * T_TOK + ridx];
  const int t = ent >> 4, slot = ent & 15;
  const float g = gates[t * H_TOP + slot];
  const bf16_t* arow = ob + ((size_t)t * H_TOP + slot) * D_DIM;
  const v16bf a0 = load_a_frag(arow + 0, half);
  const v16bf a1 = load_a_frag(arow + 32, half);
  const bf16_t* wbase = woutT + (size_t)e * C_DIM * D_DIM;

  for (int nt = wave; nt < C_DIM / 16; nt += 4) {
    const bf16_t* bcol = wbase + (size_t)(nt * 16 + n16) * D_DIM;
    v8f c = vzero8();
    c = wmma_bf16(a0, load_b_frag(bcol + 0, half), c);
    c = wmma_bf16(a1, load_b_frag(bcol + 32, half), c);
#pragma unroll
    for (int r = 0; r < 8; ++r) {
      const int M = r + 8 * half;
      const int tS = __shfl(t, M, 32);
      const float gS = __shfl(g, M, 32);
      const int col = nt * 16 + n16;
      float val = gS * (c[r] + b_out[e * C_DIM + col]);
      if (base + M < cnt) unsafeAtomicAdd(&out[(size_t)tS * C_DIM + col], val);
    }
  }
}

// ---------------------------------------------------------------- host launch
extern "C" void kernel_launch(void* const* d_in, const int* in_sizes, int n_in,
                              void* d_out, int out_size, void* d_ws, size_t ws_size,
                              hipStream_t stream) {
  (void)in_sizes; (void)n_in; (void)out_size; (void)ws_size;
  const float* x = (const float*)d_in[0];
  const float* wgate = (const float*)d_in[1];
  const float* b_in = (const float*)d_in[3];
  const float* W_out = (const float*)d_in[4];
  const float* b_out = (const float*)d_in[5];
  const float* W_kv = (const float*)d_in[6];
  const float* b_kv = (const float*)d_in[7];
  const float* W_in = (const float*)d_in[2];
  float* out = (float*)d_out;

  char* ws = (char*)d_ws;
  size_t off = 0;
  auto take = [&](size_t bytes) {
    char* p = ws + off;
    off += (bytes + 255) & ~(size_t)255;
    return p;
  };
  bf16_t* x_bf = (bf16_t*)take((size_t)T_TOK * C_DIM * 2);
  bf16_t* wkvT = (bf16_t*)take((size_t)2 * D_DIM * C_DIM * 2);
  bf16_t* winT = (bf16_t*)take((size_t)E_EXP * D_DIM * C_DIM * 2);
  bf16_t* woutT = (bf16_t*)take((size_t)E_EXP * C_DIM * D_DIM * 2);
  bf16_t* k_bf = (bf16_t*)take((size_t)T_TOK * D_DIM * 2);
  bf16_t* vt_bf = (bf16_t*)take((size_t)T_TOK * D_DIM * 2);
  bf16_t* q_bf = (bf16_t*)take((size_t)T_TOK * H_TOP * D_DIM * 2);
  bf16_t* o_bf = (bf16_t*)take((size_t)T_TOK * H_TOP * D_DIM * 2);
  float* gates = (float*)take((size_t)T_TOK * H_TOP * 4);
  int* entries = (int*)take((size_t)E_EXP * T_TOK * 4);
  int* counts = (int*)take(E_EXP * 4);
  float* p_sum = (float*)take(E_EXP * 4);
  float* z_acc = (float*)take(4);

  zero_kernel<<<(T_TOK * C_DIM) / 256, 256, 0, stream>>>(out, counts, p_sum, z_acc);
  cvt_x_kernel<<<(T_TOK * C_DIM) / 256, 256, 0, stream>>>(x, x_bf);
  cvt_wkv_kernel<<<(2 * D_DIM * C_DIM) / 256, 256, 0, stream>>>(W_kv, wkvT);
  cvt_win_kernel<<<(E_EXP * D_DIM * C_DIM) / 256, 256, 0, stream>>>(W_in, winT);
  cvt_wout_kernel<<<(E_EXP * C_DIM * D_DIM) / 256, 256, 0, stream>>>(W_out, woutT);
  gate_kernel<<<T_TOK / 4, 128, 0, stream>>>(x, wgate, gates, entries, counts, p_sum,
                                             z_acc);
  aux_kernel<<<1, 32, 0, stream>>>(p_sum, counts, z_acc, out + (size_t)T_TOK * C_DIM);
  kv_kernel<<<(T_TOK / 16) * 8, 32, 0, stream>>>(x_bf, wkvT, b_kv, k_bf, vt_bf);
  qproj_kernel<<<dim3(E_EXP, T_TOK / 16), 32, 0, stream>>>(x_bf, winT, b_in, entries,
                                                           counts, q_bf);
  attn_kernel<<<(2 * H_TOP * (N_SEQ / 16)) / 4, 128, 0, stream>>>(q_bf, k_bf, vt_bf,
                                                                  o_bf);
  oproj_kernel<<<dim3(E_EXP, T_TOK / 16), 128, 0, stream>>>(o_bf, woutT, b_out, gates,
                                                            entries, counts, out);
}
